// OLSTMModel_3882650435794
// MI455X (gfx1250) — matrix operand order repaired and checked by reading, not compiled, not run on hardware
//
#include <hip/hip_runtime.h>
#include <hip/hip_bf16.h>

// ---------------- problem constants (from reference) ----------------
#define T_STEPS 20
#define N_NODES 50000
#define K_ACT   10000      // 625 * 16, exactly tileable
#define RNN     128
#define EMB     64
#define G2      16
#define OUTSZ   5
#define KDIM    256        // concat: [ie(64), te(64), h(128)]
#define GATES   512        // 4*RNN

typedef __attribute__((ext_vector_type(16))) __bf16 v16bf;
typedef __attribute__((ext_vector_type(8)))  float  v8f;

__device__ __forceinline__ __bf16 f2bf(float f) {
    union { float f; unsigned u; } in; in.f = f;
    unsigned u = in.u;
    u += 0x7FFFu + ((u >> 16) & 1u);            // round to nearest even
    union { unsigned short s; __bf16 b; } out; out.s = (unsigned short)(u >> 16);
    return out.b;
}

__device__ __forceinline__ float sigmoidf(float x) { return 1.0f / (1.0f + __expf(-x)); }
__device__ __forceinline__ float tanh_fast(float x) { return 2.0f * sigmoidf(2.0f * x) - 1.0f; }

// ---------------- init: zero outputs region, copy h0/c0 into state ----------------
__global__ void k_init(float* __restrict__ out, const float* __restrict__ h0,
                       const float* __restrict__ c0) {
    const long OUT_ELEMS = (long)T_STEPS * N_NODES * OUTSZ;   // 5,000,000
    const long HC        = (long)N_NODES * RNN;               // 6,400,000
    float* h = out + OUT_ELEMS;
    float* c = h + HC;
    long i = (long)blockIdx.x * blockDim.x + threadIdx.x;
    long stride = (long)gridDim.x * blockDim.x;
    for (long x = i; x < OUT_ELEMS; x += stride) out[x] = 0.0f;
    for (long x = i; x < HC; x += stride) { h[x] = h0[x]; c[x] = c0[x]; }
}

// ---------------- pack Wc = [W_ih ; W_hh] into WMMA B fragments + combined bias --------
// B fragment layout as packed here (contiguous k order, same as the A loads):
//   lane l: column N = n0 + (l&15); K rows = kk + (l>>4)*16 + e, e=0..15 (2 bf16/VGPR)
// Fragments stored contiguously: frag(ntg, kblk)[lane][e], 32 ntg x 8 kblk x 32 x 16.
__global__ void k_prep(__bf16* __restrict__ wcfrag, float* __restrict__ bc,
                       const float* __restrict__ W_ih, const float* __restrict__ W_hh,
                       const float* __restrict__ b_ih, const float* __restrict__ b_hh) {
    int tid = blockIdx.x * blockDim.x + threadIdx.x;   // 131072 threads
    if (tid < GATES) bc[tid] = b_ih[tid] + b_hh[tid];
    if (tid >= 32 * 8 * 32 * 16) return;
    int e    = tid & 15;
    int lane = (tid >> 4) & 31;
    int kblk = (tid >> 9) & 7;
    int ntg  = tid >> 12;                               // 0..31 (global N tile over 512)
    int kr = kblk * 32 + ((lane >> 4) << 4) + e;        // K row in [0,256)
    int nc = ntg * 16 + (lane & 15);                    // col in [0,512)
    float v = (kr < 2 * EMB) ? W_ih[kr * GATES + nc] : W_hh[(kr - 2 * EMB) * GATES + nc];
    wcfrag[tid] = f2bf(v);
}

// ---------------- embed + gather: build A[k, 0:256] = [relu(x@Win+b), relu(obs@Wobs+b), h]
__global__ void k_embed(int t, __bf16* __restrict__ A,
                        const float* __restrict__ input_data, const float* __restrict__ grids,
                        const int* __restrict__ active_idx, const float* __restrict__ h_state,
                        const float* __restrict__ W_in, const float* __restrict__ b_in,
                        const float* __restrict__ W_obs, const float* __restrict__ b_obs) {
    int k = blockIdx.x * blockDim.x + threadIdx.x;
    if (k >= K_ACT) return;
    int idx = active_idx[t * K_ACT + k];
    float x0 = input_data[((long)t * N_NODES + idx) * 2 + 0];
    float x1 = input_data[((long)t * N_NODES + idx) * 2 + 1];
    __bf16* arow = A + (long)k * KDIM;
    #pragma unroll 8
    for (int e = 0; e < EMB; ++e) {
        float v = fmaf(x0, W_in[e], fmaf(x1, W_in[EMB + e], b_in[e]));
        arow[e] = f2bf(fmaxf(v, 0.0f));
    }
    const float* obs = grids + ((long)t * K_ACT + k) * G2;
    float o[G2];
    #pragma unroll
    for (int r = 0; r < G2; ++r) o[r] = obs[r];
    for (int e = 0; e < EMB; ++e) {
        float v = b_obs[e];
        #pragma unroll
        for (int r = 0; r < G2; ++r) v = fmaf(o[r], W_obs[r * EMB + e], v);
        arow[EMB + e] = f2bf(fmaxf(v, 0.0f));
    }
    const float* hrow = h_state + (long)idx * RNN;
    #pragma unroll 8
    for (int j = 0; j < RNN; ++j) arow[2 * EMB + j] = f2bf(hrow[j]);
}

// ---------------- GEMM + fused LSTM elementwise ----------------
// One wave computes a 16-row x 32-col region: TWO 16-col tiles (nt, nt+4) across
// ALL FOUR gates -> 8 accumulators, 64 x wmma_f32_16x16x32_bf16 per wave.
// Each A-fragment load feeds 8 WMMAs; B fragments stream from L2-resident wcfrag.
// Epilogue: bias, sigmoid/tanh, c_new/h_new -> compact Hnew/Cnew buffers.
__global__ void __launch_bounds__(32) k_gemm(int t,
        const __bf16* __restrict__ A, const __bf16* __restrict__ wcfrag,
        const float* __restrict__ bc, const int* __restrict__ active_idx,
        const float* __restrict__ c_state,
        float* __restrict__ Hnew, float* __restrict__ Cnew) {
    const int lane   = threadIdx.x;      // 0..31
    const int ntile0 = blockIdx.x;       // 0..3 ; second tile = ntile0 + 4
    const int mtile  = blockIdx.y;       // 0..624
    const int m0 = mtile * 16;

    const v8f z = {0.f, 0.f, 0.f, 0.f, 0.f, 0.f, 0.f, 0.f};
    v8f acc[2][4];
    #pragma unroll
    for (int c = 0; c < 2; ++c)
        #pragma unroll
        for (int g = 0; g < 4; ++g) acc[c][g] = z;

    // A fragment addressing: lane l holds row M = m0+(l&15),
    // elements K = kk + (l>>4)*16 + {0..15}
    const int mA   = m0 + (lane & 15);
    const int koff = (lane >> 4) << 4;
    const __bf16* arow = A + (long)mA * KDIM + koff;
    const long ntstride = (long)8 * 32 * 16;     // elements per global N-tile in wcfrag

    #pragma unroll
    for (int kk = 0; kk < KDIM; kk += 32) {
        union { v16bf v; uint4 u[2]; } af;
        af.u[0] = *reinterpret_cast<const uint4*>(arow + kk);
        af.u[1] = *reinterpret_cast<const uint4*>(arow + kk + 8);
        const int kblk = kk >> 5;
        const __bf16* fb = wcfrag + ((long)kblk * 32 + lane) * 16;
        #pragma unroll
        for (int c = 0; c < 2; ++c) {
            const int nt = ntile0 + c * 4;       // 16-col tile within one gate's 128 cols
            #pragma unroll
            for (int g = 0; g < 4; ++g) {
                const v16bf bfr = *reinterpret_cast<const v16bf*>(
                    fb + (long)(g * 8 + nt) * ntstride);
                acc[c][g] = __builtin_amdgcn_wmma_f32_16x16x32_bf16(
                    false, af.v, false, bfr, (short)0, acc[c][g], false, false);
            }
        }
    }

    // D layout: lane l, vgpr v -> row M = m0 + (l>>4)*8 + v, col = n0 + (l&15)
    #pragma unroll
    for (int c = 0; c < 2; ++c) {
        const int nt = ntile0 + c * 4;
        const int j  = nt * 16 + (lane & 15);    // col within gate, [0,128)
        const float bi  = bc[j];
        const float bf_ = bc[RNN + j];
        const float bg  = bc[2 * RNN + j];
        const float bo  = bc[3 * RNN + j];
        #pragma unroll
        for (int v = 0; v < 8; ++v) {
            const int m = m0 + ((lane >> 4) << 3) + v;
            const int idx = active_idx[t * K_ACT + m];
            const float gi = acc[c][0][v] + bi;
            const float gf = acc[c][1][v] + bf_;
            const float gg = acc[c][2][v] + bg;
            const float go = acc[c][3][v] + bo;
            const float c_old = c_state[(long)idx * RNN + j];
            const float cn = sigmoidf(gf) * c_old + sigmoidf(gi) * tanh_fast(gg);
            const float hn = sigmoidf(go) * tanh_fast(cn);
            Cnew[(long)m * RNN + j] = cn;
            Hnew[(long)m * RNN + j] = hn;
        }
    }
}

// ---------------- output projection + scatter (1 wave per ped) ----------------
__global__ void k_out(int t, const float* __restrict__ Hnew, const float* __restrict__ Cnew,
                      const int* __restrict__ active_idx,
                      const float* __restrict__ W_out, const float* __restrict__ b_out,
                      float* __restrict__ out, float* __restrict__ h_state,
                      float* __restrict__ c_state) {
    const int wave = threadIdx.x >> 5;
    const int lane = threadIdx.x & 31;
    const int k = blockIdx.x * (blockDim.x >> 5) + wave;
    if (k >= K_ACT) return;
    const int idx = active_idx[t * K_ACT + k];
    const float* hr = Hnew + (long)k * RNN;
    const float* cr = Cnew + (long)k * RNN;
    float p[OUTSZ] = {0.f, 0.f, 0.f, 0.f, 0.f};
    #pragma unroll
    for (int r = lane; r < RNN; r += 32) {
        const float hv = hr[r];
        #pragma unroll
        for (int jj = 0; jj < OUTSZ; ++jj) p[jj] = fmaf(hv, W_out[r * OUTSZ + jj], p[jj]);
        h_state[(long)idx * RNN + r] = hv;           // scatter new h
        c_state[(long)idx * RNN + r] = cr[r];        // scatter new c
    }
    #pragma unroll
    for (int jj = 0; jj < OUTSZ; ++jj)
        for (int off = 16; off >= 1; off >>= 1)
            p[jj] += __shfl_down(p[jj], off, 32);
    if (lane == 0) {
        float* orow = out + ((long)t * N_NODES + idx) * OUTSZ;
        #pragma unroll
        for (int jj = 0; jj < OUTSZ; ++jj) orow[jj] = p[jj] + b_out[jj];
    }
}

// ---------------- host launch ----------------
extern "C" void kernel_launch(void* const* d_in, const int* in_sizes, int n_in,
                              void* d_out, int out_size, void* d_ws, size_t ws_size,
                              hipStream_t stream) {
    (void)in_sizes; (void)n_in; (void)out_size; (void)ws_size;
    const float* input_data = (const float*)d_in[0];
    const float* grids      = (const float*)d_in[1];
    const float* h0         = (const float*)d_in[2];
    const float* c0         = (const float*)d_in[3];
    const int*   active_idx = (const int*)  d_in[4];
    const float* W_in       = (const float*)d_in[5];
    const float* b_in       = (const float*)d_in[6];
    const float* W_obs      = (const float*)d_in[7];
    const float* b_obs      = (const float*)d_in[8];
    const float* W_ih       = (const float*)d_in[9];
    const float* b_ih       = (const float*)d_in[10];
    const float* W_hh       = (const float*)d_in[11];
    const float* b_hh       = (const float*)d_in[12];
    const float* W_out      = (const float*)d_in[13];
    const float* b_out      = (const float*)d_in[14];

    float* out     = (float*)d_out;
    float* h_state = out + (size_t)T_STEPS * N_NODES * OUTSZ;   // reuse h_fin region as state
    float* c_state = h_state + (size_t)N_NODES * RNN;           // reuse c_fin region as state

    char* ws = (char*)d_ws;
    size_t off = 0;
    __bf16* wcfrag = (__bf16*)(ws + off); off += (size_t)32 * 8 * 32 * 16 * 2;   // 256 KB
    float*  bc     = (float*)(ws + off);  off += (size_t)GATES * 4;
    off = (off + 255) & ~(size_t)255;
    __bf16* A      = (__bf16*)(ws + off); off += (size_t)K_ACT * KDIM * 2;       // 5.12 MB
    off = (off + 255) & ~(size_t)255;
    float*  Hnew   = (float*)(ws + off);  off += (size_t)K_ACT * RNN * 4;        // 5.12 MB
    float*  Cnew   = (float*)(ws + off);  off += (size_t)K_ACT * RNN * 4;        // 5.12 MB

    k_init<<<2048, 256, 0, stream>>>(out, h0, c0);
    k_prep<<<512, 256, 0, stream>>>(wcfrag, bc, W_ih, W_hh, b_ih, b_hh);

    const dim3 ggrid(4, K_ACT / 16);   // 4 blocks x 2 col tiles each, 625 row tiles, 1 wave
    for (int t = 0; t < T_STEPS; ++t) {
        k_embed<<<(K_ACT + 127) / 128, 128, 0, stream>>>(t, A, input_data, grids, active_idx,
                                                         h_state, W_in, b_in, W_obs, b_obs);
        k_gemm<<<ggrid, 32, 0, stream>>>(t, A, wcfrag, bc, active_idx, c_state, Hnew, Cnew);
        k_out<<<(K_ACT + 7) / 8, 256, 0, stream>>>(t, Hnew, Cnew, active_idx, W_out, b_out,
                                                   out, h_state, c_state);
    }
}
